// MoEFFN_76192719831540
// MI455X (gfx1250) — matrix operand order repaired and checked by reading, not compiled, not run on hardware
//
#include <hip/hip_runtime.h>
#include <hip/hip_bf16.h>

// ---------------------------------------------------------------------------
// MoE FFN (shared expert + top-4 of 16 routed experts), bf16 WMMA pipeline
// Shapes: B=4, T=2048 -> N=8192 tokens, C=768, H=3072, E=16, K=4
// CDNA5 paths: v_wmma_f32_16x16x32_bf16, global_load_async_to_lds_b128
// (double-buffered, s_wait_asynccnt 1 pipelining), global_prefetch
// ---------------------------------------------------------------------------

typedef __attribute__((ext_vector_type(16))) __bf16 v16bf;
typedef __attribute__((ext_vector_type(8)))  __bf16 bf16x8;
typedef __attribute__((ext_vector_type(8)))  float  v8f;

#define N_TOK 8192
#define C_DIM 768
#define H_DIM 3072
#define E_NUM 16
#define TOPK  4

#define BM 64        // token rows per block tile
#define BN 64        // output cols per block tile
#define BK 32        // K chunk (wmma bf16 K=32)
#define A_PITCH 40   // halfwords per A row (80B, keeps 16B chunks aligned)
#define B_PITCH 40   // halfwords per transposed B column

// ---- CDNA5 async global->LDS staging -------------------------------------
__device__ __forceinline__ void async_b128(unsigned lds_off, const void* gaddr) {
  asm volatile("global_load_async_to_lds_b128 %0, %1, off"
               :: "v"(lds_off), "v"(gaddr) : "memory");
}
__device__ __forceinline__ void wait_async0() {
#if __has_builtin(__builtin_amdgcn_s_wait_asynccnt)
  __builtin_amdgcn_s_wait_asynccnt(0);
#else
  asm volatile("s_wait_asynccnt 0x0" ::: "memory");
#endif
}
__device__ __forceinline__ void wait_async1() {   // allow 1 in flight (next tile)
#if __has_builtin(__builtin_amdgcn_s_wait_asynccnt)
  __builtin_amdgcn_s_wait_asynccnt(1);
#else
  asm volatile("s_wait_asynccnt 0x1" ::: "memory");
#endif
}
// Generic pointers to __shared__ carry the LDS offset in their low 32 bits.
__device__ __forceinline__ unsigned lds_off(const void* p) {
  return (unsigned)(size_t)p;
}

__device__ __forceinline__ v16bf frag_from2(bf16x8 lo, bf16x8 hi) {
  return __builtin_shufflevector(lo, hi, 0, 1, 2, 3, 4, 5, 6, 7,
                                 8, 9, 10, 11, 12, 13, 14, 15);
}

// ---------------------------------------------------------------------------
// fp32 -> bf16 conversion (grid-stride, 4 elems/thread/iter)
// ---------------------------------------------------------------------------
struct bf4 { __bf16 a, b, c, d; };

__global__ void cvt_bf16_kernel(const float* __restrict__ src,
                                __bf16* __restrict__ dst, long n4) {
  long i = (long)blockIdx.x * blockDim.x + threadIdx.x;
  long stride = (long)gridDim.x * blockDim.x;
  for (; i < n4; i += stride) {
    long j = i * 4;
    float4 f = *(const float4*)(src + j);
    bf4 o;
    o.a = (__bf16)f.x; o.b = (__bf16)f.y; o.c = (__bf16)f.z; o.d = (__bf16)f.w;
    *(bf4*)(dst + j) = o;
  }
}

// ---------------------------------------------------------------------------
// Gating: one wave per token; compacted per-expert (token, weight) lists.
// ---------------------------------------------------------------------------
__global__ __launch_bounds__(256)
void gate_kernel(const float* __restrict__ x,
                 const float* __restrict__ w_gate,     // [C][E]
                 const float* __restrict__ bias,       // [E]
                 int*   __restrict__ counts,           // [E]
                 int*   __restrict__ lidx,             // [E][N]
                 float* __restrict__ lw) {             // [E][N]
  __shared__ float wg[E_NUM * C_DIM];                  // transposed: [e][c]
  for (int t = threadIdx.x; t < E_NUM * C_DIM; t += blockDim.x) {
    int e = t / C_DIM, c = t - e * C_DIM;
    wg[t] = w_gate[c * E_NUM + e];
  }
  __syncthreads();

  int wave = threadIdx.x >> 5, lane = threadIdx.x & 31;
  int token = blockIdx.x * 8 + wave;
  if (token >= N_TOK) return;

  float acc[E_NUM];
#pragma unroll
  for (int e = 0; e < E_NUM; ++e) acc[e] = 0.f;

  const float* xr = x + (long)token * C_DIM;
  for (int c = lane; c < C_DIM; c += 32) {
    float xv = xr[c];
#pragma unroll
    for (int e = 0; e < E_NUM; ++e) acc[e] += xv * wg[e * C_DIM + c];
  }
#pragma unroll
  for (int off = 16; off > 0; off >>= 1) {
#pragma unroll
    for (int e = 0; e < E_NUM; ++e) acc[e] += __shfl_xor(acc[e], off, 32);
  }

  if (lane == 0) {
    float g[E_NUM];
#pragma unroll
    for (int e = 0; e < E_NUM; ++e)
      g[e] = 1.f / (1.f + expf(-(acc[e] + bias[e])));
    int   pick[TOPK];
    float pw[TOPK];
    float wsum = 0.f;
#pragma unroll
    for (int k = 0; k < TOPK; ++k) {
      int   bi = -1;
      float bv = -1.f;
#pragma unroll
      for (int e = 0; e < E_NUM; ++e) {
        bool taken = false;
        for (int q = 0; q < k; ++q) taken |= (pick[q] == e);
        if (!taken && g[e] > bv) { bv = g[e]; bi = e; }
      }
      pick[k] = bi; pw[k] = bv; wsum += bv;
    }
    float inv = 1.f / wsum;
#pragma unroll
    for (int k = 0; k < TOPK; ++k) {
      int e = pick[k];
      int slot = atomicAdd(&counts[e], 1);
      lidx[e * N_TOK + slot] = token;
      lw  [e * N_TOK + slot] = pw[k] * inv;
    }
  }
}

// ---------------------------------------------------------------------------
// Pass A: act[m, :H] = swiglu(x_gathered[m, :C] @ wfc[:C, :2H])   (bf16 WMMA)
// 8 waves: wm in {0,1} x wn in {0..3}; each wave owns a 32x16 strip.
// Double-buffered LDS; A via async DMA one tile ahead (s_wait_asynccnt 1);
// B pipelined through registers one tile ahead.
// ---------------------------------------------------------------------------
__global__ __launch_bounds__(256)
void ffn_fc_kernel(const __bf16* __restrict__ xb,     // [N][C]
                   const __bf16* __restrict__ wfc,    // [C][2H] (this expert)
                   const int*    __restrict__ idx,    // null => identity
                   const int*    __restrict__ cnt,    // null => N
                   __bf16*       __restrict__ act) {  // [N][H] compacted rows
  const int count = cnt ? cnt[0] : N_TOK;
  const int m0 = blockIdx.x * BM;
  if (m0 >= count) return;
  const int n0 = blockIdx.y * BN;

  __shared__ __bf16 sA [2][BM * A_PITCH];
  __shared__ __bf16 sBg[2][BN * B_PITCH];   // transposed: [col][k]
  __shared__ __bf16 sBv[2][BN * B_PITCH];
  __shared__ int    sTok[BM];

  const int tid = threadIdx.x;
  if (tid < BM) {
    int m = m0 + tid;
    sTok[tid] = (m < count) ? (idx ? idx[m] : m) : 0;
  }
  __syncthreads();

  // --- staging maps ---
  const int arow = tid >> 2;                 // 0..63
  const int ac8  = (tid & 3) * 8;            // 0,8,16,24
  const long abase = (long)sTok[arow] * C_DIM + ac8;
  const unsigned aoff0 = lds_off(&sA[0][arow * A_PITCH + ac8]);
  const unsigned aoff1 = lds_off(&sA[1][arow * A_PITCH + ac8]);
  const int bk  = tid >> 3;                  // 0..31 (k row)
  const int bc8 = (tid & 7) * 8;             // col chunk
  const __bf16* bsrc = wfc + (long)bk * (2 * H_DIM) + n0 + bc8;

  const int lane = tid & 31, wave = tid >> 5;
  const int wm = wave >> 2, wn = wave & 3;
  const int fr  = (lane & 15);               // frag row/col within 16
  const int ks  = (lane >> 4) * 8;           // A: K start {0,8}
  const int klo = (lane >> 4) * 8;           // B: K start {0,8}

  v8f cg0 = {}, cg1 = {}, cv0 = {}, cv1 = {};

  const int ITERS = C_DIM / BK;              // 24

  // ---- prologue: stage tile 0 ----
  async_b128(aoff0, xb + abase);
  bf16x8 rg = *(const bf16x8*)bsrc;
  bf16x8 rv = *(const bf16x8*)(bsrc + H_DIM);

  for (int it = 0; it < ITERS; ++it) {
    const int cur = it & 1;
    const long knext = (long)(it + 1) * BK;
    __syncthreads();                         // reads of buf `cur` (it-2) done
    // ---- commit B(it) registers into LDS (transposed) ----
#pragma unroll
    for (int j = 0; j < 8; ++j) {
      sBg[cur][(bc8 + j) * B_PITCH + bk] = rg[j];
      sBv[cur][(bc8 + j) * B_PITCH + bk] = rv[j];
    }
    // ---- issue tile it+1 (A async, B global loads into registers) ----
    if (it + 1 < ITERS) {
      async_b128(cur ? aoff0 : aoff1, xb + abase + knext);
      rg = *(const bf16x8*)(bsrc + knext * (2 * H_DIM));
      rv = *(const bf16x8*)(bsrc + knext * (2 * H_DIM) + H_DIM);
      __builtin_prefetch(wfc + (knext + BK) * (2 * H_DIM) + n0, 0, 1);
      wait_async1();                         // A(it) resident, A(it+1) in flight
    } else {
      wait_async0();
    }
    __syncthreads();                         // all waves staged tile `it`

    // ---- fragments + 4 WMMAs ----
    const __bf16* pb_g = &sBg[cur][(wn * 16 + fr) * B_PITCH + klo];
    const __bf16* pb_v = &sBv[cur][(wn * 16 + fr) * B_PITCH + klo];
    v16bf fbg = frag_from2(*(const bf16x8*)pb_g, *(const bf16x8*)(pb_g + 16));
    v16bf fbv = frag_from2(*(const bf16x8*)pb_v, *(const bf16x8*)(pb_v + 16));

    const __bf16* pa0 = &sA[cur][(wm * 32 + fr) * A_PITCH + ks];
    const __bf16* pa1 = &sA[cur][(wm * 32 + 16 + fr) * A_PITCH + ks];
    v16bf fa0 = frag_from2(*(const bf16x8*)pa0, *(const bf16x8*)(pa0 + 16));
    v16bf fa1 = frag_from2(*(const bf16x8*)pa1, *(const bf16x8*)(pa1 + 16));

    cg0 = __builtin_amdgcn_wmma_f32_16x16x32_bf16(false, fa0, false, fbg,
                                                  (short)0, cg0, false, false);
    cg1 = __builtin_amdgcn_wmma_f32_16x16x32_bf16(false, fa1, false, fbg,
                                                  (short)0, cg1, false, false);
    cv0 = __builtin_amdgcn_wmma_f32_16x16x32_bf16(false, fa0, false, fbv,
                                                  (short)0, cv0, false, false);
    cv1 = __builtin_amdgcn_wmma_f32_16x16x32_bf16(false, fa1, false, fbv,
                                                  (short)0, cv1, false, false);
  }

  // ---- SwiGLU + store bf16 activations ----
  const int col = n0 + wn * 16 + fr;
#pragma unroll
  for (int sub = 0; sub < 2; ++sub) {
    v8f g8 = sub ? cg1 : cg0;
    v8f u8 = sub ? cv1 : cv0;
#pragma unroll
    for (int i = 0; i < 8; ++i) {
      int r = wm * 32 + sub * 16 + i + ((lane >> 4) << 3);
      int m = m0 + r;
      if (m < count) {
        float g = g8[i], v = u8[i];
        float s = g / (1.f + expf(-g));      // silu(g)
        act[(long)m * H_DIM + col] = (__bf16)(s * v);
      }
    }
  }
}

// ---------------------------------------------------------------------------
// Pass B: out[token] (+)= w[m] * (act[m, :H] @ wproj[:H, :C])     (bf16 WMMA)
// ---------------------------------------------------------------------------
__global__ __launch_bounds__(256)
void ffn_proj_kernel(const __bf16* __restrict__ act,    // [N][H] compacted
                     const __bf16* __restrict__ wproj,  // [H][C] (this expert)
                     const int*    __restrict__ idx,    // null => identity
                     const float*  __restrict__ wts,    // null => 1.0
                     const int*    __restrict__ cnt,    // null => N
                     float*        __restrict__ out,    // [N][C]
                     int add_mode) {
  const int count = cnt ? cnt[0] : N_TOK;
  const int m0 = blockIdx.x * BM;
  if (m0 >= count) return;
  const int n0 = blockIdx.y * BN;

  __shared__ __bf16 sA[2][BM * A_PITCH];
  __shared__ __bf16 sB[2][BN * B_PITCH];    // transposed: [col][k]
  __shared__ int    sTok[BM];
  __shared__ float  sW[BM];

  const int tid = threadIdx.x;
  if (tid < BM) {
    int m = m0 + tid;
    if (m < count) {
      sTok[tid] = idx ? idx[m] : m;
      sW[tid]   = wts ? wts[m] : 1.f;
    } else { sTok[tid] = 0; sW[tid] = 0.f; }
  }
  __syncthreads();

  const int arow = tid >> 2;
  const int ac8  = (tid & 3) * 8;
  const long abase = (long)(m0 + arow) * H_DIM + ac8;   // compact rows
  const unsigned aoff0 = lds_off(&sA[0][arow * A_PITCH + ac8]);
  const unsigned aoff1 = lds_off(&sA[1][arow * A_PITCH + ac8]);
  const int bk  = tid >> 3;
  const int bc8 = (tid & 7) * 8;
  const __bf16* bsrc = wproj + (long)bk * C_DIM + n0 + bc8;

  const int lane = tid & 31, wave = tid >> 5;
  const int wm = wave >> 2, wn = wave & 3;
  const int fr  = (lane & 15);
  const int ks  = (lane >> 4) * 8;
  const int klo = (lane >> 4) * 8;

  v8f c0 = {}, c1 = {};

  const int ITERS = H_DIM / BK;              // 96

  async_b128(aoff0, act + abase);
  bf16x8 rb = *(const bf16x8*)bsrc;

  for (int it = 0; it < ITERS; ++it) {
    const int cur = it & 1;
    const long knext = (long)(it + 1) * BK;
    __syncthreads();
#pragma unroll
    for (int j = 0; j < 8; ++j)
      sB[cur][(bc8 + j) * B_PITCH + bk] = rb[j];
    if (it + 1 < ITERS) {
      async_b128(cur ? aoff0 : aoff1, act + abase + knext);
      rb = *(const bf16x8*)(bsrc + knext * C_DIM);
      __builtin_prefetch(wproj + (knext + BK) * C_DIM + n0, 0, 1);
      wait_async1();
    } else {
      wait_async0();
    }
    __syncthreads();

    const __bf16* pb = &sB[cur][(wn * 16 + fr) * B_PITCH + klo];
    v16bf fb = frag_from2(*(const bf16x8*)pb, *(const bf16x8*)(pb + 16));

    const __bf16* pa0 = &sA[cur][(wm * 32 + fr) * A_PITCH + ks];
    const __bf16* pa1 = &sA[cur][(wm * 32 + 16 + fr) * A_PITCH + ks];
    v16bf fa0 = frag_from2(*(const bf16x8*)pa0, *(const bf16x8*)(pa0 + 16));
    v16bf fa1 = frag_from2(*(const bf16x8*)pa1, *(const bf16x8*)(pa1 + 16));

    c0 = __builtin_amdgcn_wmma_f32_16x16x32_bf16(false, fa0, false, fb,
                                                 (short)0, c0, false, false);
    c1 = __builtin_amdgcn_wmma_f32_16x16x32_bf16(false, fa1, false, fb,
                                                 (short)0, c1, false, false);
  }

  const int col = n0 + wn * 16 + fr;
#pragma unroll
  for (int sub = 0; sub < 2; ++sub) {
    v8f cc = sub ? c1 : c0;
#pragma unroll
    for (int i = 0; i < 8; ++i) {
      int r = wm * 32 + sub * 16 + i + ((lane >> 4) << 3);
      int m = m0 + r;
      if (m < count) {
        float val = cc[i] * sW[r];
        float* dst = &out[(long)sTok[r] * C_DIM + col];
        if (add_mode) atomicAdd(dst, val);
        else          *dst = val;
      }
    }
  }
}

// ---------------------------------------------------------------------------
// Host launcher
// ---------------------------------------------------------------------------
extern "C" void kernel_launch(void* const* d_in, const int* in_sizes, int n_in,
                              void* d_out, int out_size, void* d_ws, size_t ws_size,
                              hipStream_t stream) {
  (void)in_sizes; (void)n_in; (void)out_size; (void)ws_size;

  const float* x        = (const float*)d_in[0];
  const float* wsfc     = (const float*)d_in[1];
  const float* wsproj   = (const float*)d_in[2];
  const float* wefc     = (const float*)d_in[3];
  const float* weproj   = (const float*)d_in[4];
  const float* w_gate   = (const float*)d_in[5];
  const float* ebias    = (const float*)d_in[6];
  float* out            = (float*)d_out;

  char* p = (char*)d_ws;
  auto carve = [&](size_t bytes) -> void* {
    void* r = (void*)p;
    p += (bytes + 255) & ~(size_t)255;
    return r;
  };
  const size_t FC_SZ   = (size_t)C_DIM * 2 * H_DIM;   // per expert slot
  const size_t PROJ_SZ = (size_t)H_DIM * C_DIM;

  __bf16* xb        = (__bf16*)carve((size_t)N_TOK * C_DIM * 2);
  __bf16* wfc_all   = (__bf16*)carve(17 * FC_SZ * 2);
  __bf16* wproj_all = (__bf16*)carve(17 * PROJ_SZ * 2);
  int*    counts    = (int*)   carve(E_NUM * sizeof(int));
  int*    lidx      = (int*)   carve((size_t)E_NUM * N_TOK * sizeof(int));
  float*  lw        = (float*) carve((size_t)E_NUM * N_TOK * sizeof(float));
  __bf16* act       = (__bf16*)carve((size_t)N_TOK * H_DIM * 2);

  hipMemsetAsync(counts, 0, E_NUM * sizeof(int), stream);

  auto cvt = [&](const float* s, __bf16* d, long n) {
    long n4 = n / 4;
    int blocks = (int)((n4 + 255) / 256);
    if (blocks > 16384) blocks = 16384;
    cvt_bf16_kernel<<<blocks, 256, 0, stream>>>(s, d, n4);
  };
  cvt(x,      xb,                 (long)N_TOK * C_DIM);
  cvt(wsfc,   wfc_all,            (long)FC_SZ);
  cvt(wefc,   wfc_all + FC_SZ,    (long)E_NUM * FC_SZ);
  cvt(wsproj, wproj_all,          (long)PROJ_SZ);
  cvt(weproj, wproj_all + PROJ_SZ,(long)E_NUM * PROJ_SZ);

  gate_kernel<<<N_TOK / 8, 256, 0, stream>>>(x, w_gate, ebias, counts, lidx, lw);

  const dim3 gridA(N_TOK / BM, H_DIM / BN);   // 128 x 48
  const dim3 gridB(N_TOK / BM, C_DIM / BN);   // 128 x 12

  // Shared expert: plain stores initialize the output buffer.
  ffn_fc_kernel  <<<gridA, 256, 0, stream>>>(xb, wfc_all, nullptr, nullptr, act);
  ffn_proj_kernel<<<gridB, 256, 0, stream>>>(act, wproj_all, nullptr, nullptr,
                                             nullptr, out, 0);

  // Routed experts: grouped (compacted) GEMMs, f32 atomic accumulation.
  for (int e = 0; e < E_NUM; ++e) {
    ffn_fc_kernel  <<<gridA, 256, 0, stream>>>(
        xb, wfc_all + (size_t)(1 + e) * FC_SZ,
        lidx + (size_t)e * N_TOK, counts + e, act);
    ffn_proj_kernel<<<gridB, 256, 0, stream>>>(
        act, wproj_all + (size_t)(1 + e) * PROJ_SZ,
        lidx + (size_t)e * N_TOK, lw + (size_t)e * N_TOK, counts + e,
        out, 1);
  }
}
